// CharRNN_7636451852550
// MI455X (gfx1250) — compile-verified, matrix operand order checked by reading, not run on hardware
//
#include <hip/hip_runtime.h>
#include <hip/hip_bf16.h>
#include <math.h>

// ---------------------------------------------------------------------------
// CharRNN forward for MI455X (gfx1250), wave32 + WMMA bf16 (f32 accumulate).
//   VOCAB=256, EMBED=128, HIDDEN=512, BATCH=64, SEQ=1024
// d_out = logits [64,1024,256] f32  ++  final_hidden [64,512] f32
// Workspace layout (bf16):
//   hseq   : [1025][64][512]  (hseq[0] = initial hidden; hseq[l+1] = outs[l])
//   whh_t  : [512][512]   (W_hh transposed)
//   wxh_t  : [512][128]   (W_xh transposed)
//   emb_bf : [256][128]
//   why_bf : [256][512]   (W_hy_w as-is: its rows are B-matrix columns)
// Total ~68.2 MB.
// ---------------------------------------------------------------------------

#define VOCAB  256
#define EMBED  128
#define HIDDEN 512
#define BATCH  64
#define SEQ    1024

typedef __attribute__((ext_vector_type(16))) __bf16 v16bf;
typedef __attribute__((ext_vector_type(8)))  __bf16 v8bf;
typedef __attribute__((ext_vector_type(8)))  float  v8f;

// Load one 16-bf16 WMMA operand fragment for this lane.
// ISA 7.12.2 (16-bit A 16x32): lane holds K = [kbase..kbase+7] in regs 0..3
// and K = [kbase+16..kbase+23] in regs 4..7, kbase = (lane>=16 ? 8 : 0).
// Caller passes p = row_base + k0 + kbase; two contiguous b128 loads.
__device__ __forceinline__ v16bf load_frag(const __bf16* p) {
  v8bf lo = *(const v8bf*)(p);
  v8bf hi = *(const v8bf*)(p + 16);
  v16bf f;
#pragma unroll
  for (int i = 0; i < 8; ++i) { f[i] = lo[i]; f[i + 8] = hi[i]; }
  return f;
}

// ---------------------------------------------------------------------------
// Kernel 1: precision conversion + weight transposes. Grid covers 524288 ids.
// ---------------------------------------------------------------------------
__global__ void __launch_bounds__(256) prep_kernel(
    const float* __restrict__ hidden, const float* __restrict__ emb,
    const float* __restrict__ Wxh, const float* __restrict__ Whh,
    const float* __restrict__ Why,
    __bf16* __restrict__ hseq, __bf16* __restrict__ whh_t,
    __bf16* __restrict__ wxh_t, __bf16* __restrict__ emb_bf,
    __bf16* __restrict__ why_bf)
{
  int i = blockIdx.x * blockDim.x + threadIdx.x;
  if (i < HIDDEN * HIDDEN) {                       // W_hh^T
    int n = i / HIDDEN, k = i % HIDDEN;
    whh_t[n * HIDDEN + k] = (__bf16)Whh[k * HIDDEN + n];
    return;
  }
  i -= HIDDEN * HIDDEN;
  if (i < HIDDEN * EMBED) {                        // W_xh^T
    int h = i / EMBED, e = i % EMBED;
    wxh_t[h * EMBED + e] = (__bf16)Wxh[e * HIDDEN + h];
    return;
  }
  i -= HIDDEN * EMBED;
  if (i < VOCAB * EMBED) { emb_bf[i] = (__bf16)emb[i]; return; }
  i -= VOCAB * EMBED;
  if (i < VOCAB * HIDDEN) { why_bf[i] = (__bf16)Why[i]; return; }
  i -= VOCAB * HIDDEN;
  if (i < BATCH * HIDDEN) { hseq[i] = (__bf16)hidden[i]; }
}

// ---------------------------------------------------------------------------
// Kernel 2: one recurrence step.
//   h_{l+1} = tanh(emb[x[:,l]] @ W_xh + h_l @ W_hh + b_h)
// 128 waves: tile grid 4 (batch) x 32 (hidden). One 16x16 tile per wave —
// this phase is serial-latency bound, so max parallel waves per step wins.
// 4 WMMA for the K=128 embedding projection + 16 WMMA for K=512 recurrence.
// ---------------------------------------------------------------------------
__global__ void __launch_bounds__(128) rnn_step_kernel(
    const int* __restrict__ x, const float* __restrict__ b_h,
    const __bf16* __restrict__ emb_bf, const __bf16* __restrict__ wxh_t,
    const __bf16* __restrict__ whh_t, __bf16* __restrict__ hseq, int l)
{
  int wave  = (blockIdx.x * blockDim.x + threadIdx.x) >> 5;   // 0..127
  int lane  = threadIdx.x & 31;
  int tileM = (wave & 3) * 16;        // batch tile
  int tileN = (wave >> 2) * 16;       // hidden tile
  int mrow  = tileM + (lane & 15);    // A row (batch index) for this lane
  int ncol  = tileN + (lane & 15);    // B column (hidden index) for this lane
  int khalf = (lane >> 4) * 8;        // K sub-offset per half-wave

  v8f acc = {};

  // Embedding-projection part: A row = embedding[x[b][l]], B = W_xh^T row.
  int token = x[mrow * SEQ + l];
  const __bf16* arow_e = emb_bf + token * EMBED;
  const __bf16* brow_e = wxh_t + ncol * EMBED;
#pragma unroll
  for (int k0 = 0; k0 < EMBED; k0 += 32) {
    v16bf a = load_frag(arow_e + k0 + khalf);
    v16bf b = load_frag(brow_e + k0 + khalf);
    acc = __builtin_amdgcn_wmma_f32_16x16x32_bf16(false, a, false, b,
                                                  (short)0, acc, false, false);
  }

  // Recurrent part: A = h_l (hseq[l]), B = W_hh^T row.
  const __bf16* arow_h = hseq + (size_t)l * (BATCH * HIDDEN) + mrow * HIDDEN;
  const __bf16* brow_h = whh_t + ncol * HIDDEN;
#pragma unroll
  for (int k0 = 0; k0 < HIDDEN; k0 += 32) {
    v16bf a = load_frag(arow_h + k0 + khalf);
    v16bf b = load_frag(brow_h + k0 + khalf);
    acc = __builtin_amdgcn_wmma_f32_16x16x32_bf16(false, a, false, b,
                                                  (short)0, acc, false, false);
  }

  // Bias + tanh, write h_{l+1} (== outs[l]) in bf16.
  float bias = b_h[ncol];
  __bf16* hout = hseq + (size_t)(l + 1) * (BATCH * HIDDEN);
#pragma unroll
  for (int i = 0; i < 8; ++i) {
    int m = tileM + i + ((lane >> 4) << 3);      // C layout: M = i + 8*(lane/16)
    float v = tanhf(acc[i] + bias);
    hout[m * HIDDEN + tileN + (lane & 15)] = (__bf16)v;
  }
}

// ---------------------------------------------------------------------------
// Kernel 3: logits = outs @ W_hy^T + b. Flat rows r = l*64 + b (65536 rows).
// outs row r lives at hseq row (r + 64).
// Register-blocked 1A x 4B: each wave computes a 16x64 output slab
// (4 accumulators), amortizing every A-fragment load across 4 WMMAs.
// 16384 waves, 64 WMMA each.
// ---------------------------------------------------------------------------
__global__ void __launch_bounds__(128) logits_kernel(
    const __bf16* __restrict__ hseq, const __bf16* __restrict__ why_bf,
    const float* __restrict__ why_b, float* __restrict__ out)
{
  int wave  = (blockIdx.x * blockDim.x + threadIdx.x) >> 5;
  int lane  = threadIdx.x & 31;
  int tileM = (wave >> 2) * 16;       // 4096 row tiles
  int tileN = (wave & 3) * 64;        // 4 slabs of 64 vocab columns
  int mrow  = tileM + (lane & 15);
  int khalf = (lane >> 4) * 8;

  const __bf16* arow = hseq + (size_t)(mrow + BATCH) * HIDDEN;        // outs[r]
  const __bf16* brow = why_bf + (size_t)(tileN + (lane & 15)) * HIDDEN;

  v8f acc[4] = {};
#pragma unroll
  for (int k0 = 0; k0 < HIDDEN; k0 += 32) {
    v16bf a = load_frag(arow + k0 + khalf);
#pragma unroll
    for (int nt = 0; nt < 4; ++nt) {
      v16bf b = load_frag(brow + (size_t)nt * 16 * HIDDEN + k0 + khalf);
      acc[nt] = __builtin_amdgcn_wmma_f32_16x16x32_bf16(
          false, a, false, b, (short)0, acc[nt], false, false);
    }
  }

#pragma unroll
  for (int nt = 0; nt < 4; ++nt) {
    int vcol = tileN + nt * 16 + (lane & 15);
    float bias = why_b[vcol];
#pragma unroll
    for (int i = 0; i < 8; ++i) {
      int r = tileM + i + ((lane >> 4) << 3);    // flat row l*64+b
      int lidx = r >> 6;                         // l
      int bidx = r & 63;                         // b
      out[((size_t)bidx * SEQ + lidx) * VOCAB + vcol] = acc[nt][i] + bias;
    }
  }
}

// ---------------------------------------------------------------------------
// Kernel 4: final hidden bf16 -> f32 into the tail of d_out.
// ---------------------------------------------------------------------------
__global__ void __launch_bounds__(256) final_hidden_kernel(
    const __bf16* __restrict__ hseq, float* __restrict__ out)
{
  int i = blockIdx.x * blockDim.x + threadIdx.x;
  if (i < BATCH * HIDDEN)
    out[i] = (float)hseq[(size_t)SEQ * (BATCH * HIDDEN) + i];
}

// ---------------------------------------------------------------------------
extern "C" void kernel_launch(void* const* d_in, const int* in_sizes, int n_in,
                              void* d_out, int out_size, void* d_ws, size_t ws_size,
                              hipStream_t stream) {
  (void)in_sizes; (void)n_in; (void)out_size; (void)ws_size;

  const int*   x      = (const int*)  d_in[0];
  const float* hidden = (const float*)d_in[1];
  const float* emb    = (const float*)d_in[2];
  const float* Wxh    = (const float*)d_in[3];
  const float* Whh    = (const float*)d_in[4];
  const float* b_h    = (const float*)d_in[5];
  const float* Why_w  = (const float*)d_in[6];
  const float* Why_b  = (const float*)d_in[7];

  __bf16* hseq   = (__bf16*)d_ws;
  __bf16* whh_t  = hseq   + (size_t)(SEQ + 1) * BATCH * HIDDEN;
  __bf16* wxh_t  = whh_t  + (size_t)HIDDEN * HIDDEN;
  __bf16* emb_bf = wxh_t  + (size_t)HIDDEN * EMBED;
  __bf16* why_bf = emb_bf + (size_t)VOCAB * EMBED;

  // 1) bf16 conversion + transposes (524288 work items).
  prep_kernel<<<2048, 256, 0, stream>>>(hidden, emb, Wxh, Whh, Why_w,
                                        hseq, whh_t, wxh_t, emb_bf, why_bf);

  // 2) 1024 sequential recurrence steps (graph nodes; kernel boundary = sync).
  for (int l = 0; l < SEQ; ++l)
    rnn_step_kernel<<<32, 128, 0, stream>>>(x, b_h, emb_bf, wxh_t, whh_t,
                                            hseq, l);

  // 3) Output projection: 16384 waves (4096 blocks x 4 waves), 16x64 per wave.
  logits_kernel<<<4096, 128, 0, stream>>>(hseq, why_bf, Why_b, (float*)d_out);

  // 4) Final hidden state.
  final_hidden_kernel<<<128, 256, 0, stream>>>(
      hseq, (float*)d_out + (size_t)BATCH * SEQ * VOCAB);
}
